// _VQLayer_81114752352931
// MI455X (gfx1250) — matrix operand order repaired and checked by reading, not compiled, not run on hardware
//
#include <hip/hip_runtime.h>

// ---------------------------------------------------------------------------
// VQ layer for MI455X (gfx1250, wave32, WMMA).
//   score(n,k) = 0.5*||e_k||^2 - z_n . e_k   (||z||^2 constant wrt argmin)
//   dot via bf16x2 split: hi*hi + hi*lo + lo*hi on v_wmma_f32_16x16x32_bf16
//   (f32 accumulate) -> ~fp32-accurate distances at bf16 matrix rate.
// Software-pipelined: double-buffered LDS emb tiles; next tile's global loads
// are issued before the current tile's WMMA chain so L2 latency is hidden.
// ---------------------------------------------------------------------------

typedef __attribute__((ext_vector_type(16))) __bf16          v16bf;
typedef __attribute__((ext_vector_type(16))) unsigned short  v16u;
typedef __attribute__((ext_vector_type(8)))  float           v8f;

#define N_TOTAL 16384
#define C_DIM   256
#define K_CODES 8192
#define HW      1024      // 32*32 spatial per image
#define M_BLOCK 128       // z rows per workgroup
#define NB      16        // codes per k-tile (per LDS buffer)
#define NT      (K_CODES / NB)   // 512 tiles
#define BUFDW   4096      // dwords per double-buffer slot (hi 2048 + lo 2048)

__device__ __forceinline__ unsigned short f2bf(float v) {
  unsigned u = __float_as_uint(v);
  u += 0x7FFFu + ((u >> 16) & 1u);        // round-to-nearest-even to bf16
  return (unsigned short)(u >> 16);
}
__device__ __forceinline__ float bf2f(unsigned short b) {
  return __uint_as_float(((unsigned)b) << 16);
}

// ---- kernel 1: half squared norms of codebook rows --------------------------
__global__ __launch_bounds__(256) void ehn_kernel(const float* __restrict__ emb,
                                                  float* __restrict__ ehn) {
  int row  = blockIdx.x * 8 + (threadIdx.x >> 5);
  int lane = threadIdx.x & 31;
  const float* p = emb + (size_t)row * C_DIM;
  float s = 0.f;
#pragma unroll
  for (int c = 0; c < C_DIM; c += 32) {
    float v = p[c + lane];
    s += v * v;
  }
#pragma unroll
  for (int m = 16; m; m >>= 1) s += __shfl_xor(s, m, 32);
  if (lane == 0) ehn[row] = 0.5f * s;
}

// ---- kernel 2: fused distance + argmin + gather -----------------------------
__global__ __launch_bounds__(256) void vq_kernel(const float* __restrict__ z,
                                                 const float* __restrict__ emb,
                                                 const float* __restrict__ ehn,
                                                 float* __restrict__ out) {
  // 32KB: phase 1 stages z (fp32 128x33 = 16.5KB), then 2 x 16KB emb-tile
  // buffers (bf16 hi/lo planes, pre-swizzled into B-fragment lane order).
  __shared__ __align__(16) unsigned int sbuf[2 * BUFDW];
  __shared__ int idx_s[M_BLOCK];

  const int tid  = threadIdx.x;
  const int w    = tid >> 5;          // wave id (0..7) -> owns rows w*16..+15
  const int lane = tid & 31;
  const int n0   = blockIdx.x * M_BLOCK;
  const int b    = n0 / HW;           // 128 | 1024: block stays in one image
  const int hw0  = n0 % HW;

  float* stagef = (float*)sbuf;

  // ---------------- Phase 1: build register-resident A fragments ------------
  // A (16x32 bf16) per wave: lanes 0-15 = rows, element i -> K = i + 16*(lane>=16) (contiguous).
  v16u a_hi[8], a_lo[8];
  const int arow = w * 16 + (lane & 15);
#pragma unroll
  for (int kc = 0; kc < 8; ++kc) {
    // stage z[:, kc*32 .. +31] for 128 rows; NCHW -> coalesced over n (== hw)
#pragma unroll
    for (int it = 0; it < 16; ++it) {
      int cl = it * 2 + (tid >> 7);   // 0..31
      int nl = tid & 127;
      float v = z[(size_t)(b * C_DIM + kc * 32 + cl) * HW + hw0 + nl];
      stagef[nl * 33 + cl] = v;       // pad 33: conflict-free
    }
    __syncthreads();
#pragma unroll
    for (int i = 0; i < 16; ++i) {
      // 16-bit A 16x32 layout (contiguous, matches B): lane<16: K=0..15; lane>=16: K=16..31
      int K = i + ((lane >= 16) ? 16 : 0);
      float v = stagef[arow * 33 + K];
      unsigned short h = f2bf(v);
      a_hi[kc][i] = h;
      a_lo[kc][i] = f2bf(v - bf2f(h));
    }
    __syncthreads();
  }

  // B-fragment store position for this thread's (code, channel-pair) slices:
  // idx2 = tid + it*256; n = idx2>>7 (0..15), c = (idx2&127)*2
  // dword off within plane = ((c>>5)*32 + lane')*8 + (c&15)/2,
  //   lane' = n + 16*((c&31)>=16)
  float best[8];
  int   bidx[8];
#pragma unroll
  for (int r = 0; r < 8; ++r) { best[r] = 3.4e38f; bidx[r] = 0; }

  // ---------------- Prologue: load tile 0 into buffer 0 ---------------------
  {
    unsigned* bhi = sbuf;
    unsigned* blo = sbuf + 2048;
#pragma unroll
    for (int it = 0; it < 8; ++it) {
      int idx2 = tid + it * 256;
      int n = idx2 >> 7;
      int c = (idx2 & 127) * 2;
      const float2 v2 = *(const float2*)(emb + (size_t)n * C_DIM + c);
      unsigned short h0 = f2bf(v2.x), h1 = f2bf(v2.y);
      unsigned short l0 = f2bf(v2.x - bf2f(h0)), l1 = f2bf(v2.y - bf2f(h1));
      int kk  = c & 31;
      int bl  = n + ((kk >= 16) ? 16 : 0);
      int off = ((c >> 5) * 32 + bl) * 8 + ((kk & 15) >> 1);
      bhi[off] = (unsigned)h0 | ((unsigned)h1 << 16);
      blo[off] = (unsigned)l0 | ((unsigned)l1 << 16);
    }
  }
  __syncthreads();

  // ---------------- Main loop: pipelined over codebook tiles ----------------
  for (int kt = 0; kt < NT; ++kt) {
    const int kbase = kt * NB;
    const int cur   = kt & 1;
    unsigned* bhi_c = sbuf + cur * BUFDW;          // compute buffer
    unsigned* blo_c = bhi_c + 2048;
    unsigned* bhi_n = sbuf + (cur ^ 1) * BUFDW;    // fill buffer
    unsigned* blo_n = bhi_n + 2048;

    // (1) issue next tile's global loads FIRST -> latency hidden under WMMA
    float2 r2[8];
    const bool have_next = (kt + 1 < NT);
    if (have_next) {
#pragma unroll
      for (int it = 0; it < 8; ++it) {
        int idx2 = tid + it * 256;
        int n = idx2 >> 7;
        int c = (idx2 & 127) * 2;
        r2[it] = *(const float2*)(emb + (size_t)(kbase + NB + n) * C_DIM + c);
      }
    }
    // per-lane half-norm for this tile (address known now -> issues early)
    float eh = ehn[kbase + (lane & 15)];
    if (kt + 2 < NT) {  // prefetch tile kt+2 (global_prefetch_b8)
      __builtin_prefetch(emb + (size_t)(kbase + 2 * NB) * C_DIM + tid * 16, 0, 1);
    }

    // (2) 24 chained WMMAs on current buffer
    v8f acc = {0.f, 0.f, 0.f, 0.f, 0.f, 0.f, 0.f, 0.f};
#pragma unroll
    for (int kc = 0; kc < 8; ++kc) {
      int boff = (kc * 32 + lane) * 8;
      v16bf bh = *(const v16bf*)(bhi_c + boff);    // 2x ds_load_b128
      v16bf bl = *(const v16bf*)(blo_c + boff);
      v16bf ah = __builtin_bit_cast(v16bf, a_hi[kc]);
      v16bf al = __builtin_bit_cast(v16bf, a_lo[kc]);
      acc = __builtin_amdgcn_wmma_f32_16x16x32_bf16(false, ah, false, bh,
                                                    (short)0, acc, false, false);
      acc = __builtin_amdgcn_wmma_f32_16x16x32_bf16(false, ah, false, bl,
                                                    (short)0, acc, false, false);
      acc = __builtin_amdgcn_wmma_f32_16x16x32_bf16(false, al, false, bh,
                                                    (short)0, acc, false, false);
    }
    // D layout: lane<16: N=lane, M=r; lane>=16: N=lane-16, M=r+8
    int code = kbase + (lane & 15);
#pragma unroll
    for (int rr = 0; rr < 8; ++rr) {
      float s = eh - acc[rr];
      if (s < best[rr]) { best[rr] = s; bidx[rr] = code; }  // '<' => 1st occ.
    }

    // (3) convert staged registers -> bf16 hi/lo planes of the other buffer
    if (have_next) {
#pragma unroll
      for (int it = 0; it < 8; ++it) {
        int idx2 = tid + it * 256;
        int n = idx2 >> 7;
        int c = (idx2 & 127) * 2;
        unsigned short h0 = f2bf(r2[it].x), h1 = f2bf(r2[it].y);
        unsigned short l0 = f2bf(r2[it].x - bf2f(h0));
        unsigned short l1 = f2bf(r2[it].y - bf2f(h1));
        int kk  = c & 31;
        int bl  = n + ((kk >= 16) ? 16 : 0);
        int off = ((c >> 5) * 32 + bl) * 8 + ((kk & 15) >> 1);
        bhi_n[off] = (unsigned)h0 | ((unsigned)h1 << 16);
        blo_n[off] = (unsigned)l0 | ((unsigned)l1 << 16);
      }
    }
    __syncthreads();   // single barrier per tile
  }

  // ---------------- Cross-lane argmin reduction (16 lanes share a row) ------
#pragma unroll
  for (int r = 0; r < 8; ++r) {
#pragma unroll
    for (int m = 1; m < 16; m <<= 1) {
      float ov = __shfl_xor(best[r], m, 32);
      int   oi = __shfl_xor(bidx[r], m, 32);
      if (ov < best[r] || (ov == best[r] && oi < bidx[r])) {
        best[r] = ov; bidx[r] = oi;
      }
    }
  }
  if (lane == 0) {
#pragma unroll
    for (int r = 0; r < 8; ++r) idx_s[w * 16 + r] = bidx[r];
  }
  if (lane == 16) {
#pragma unroll
    for (int r = 0; r < 8; ++r) idx_s[w * 16 + 8 + r] = bidx[r];
  }
  __syncthreads();

  // ---------------- Gather emb[idx] and scatter to NCHW output --------------
  const int nl    = tid & 127;
  const int chalf = tid >> 7;
  const int myidx = idx_s[nl];
#pragma unroll 4
  for (int it = 0; it < 128; ++it) {
    int c = it * 2 + chalf;
    out[(size_t)(b * C_DIM + c) * HW + hw0 + nl] =
        emb[(size_t)myidx * C_DIM + c];
  }
}

// ---------------------------------------------------------------------------
extern "C" void kernel_launch(void* const* d_in, const int* in_sizes, int n_in,
                              void* d_out, int out_size, void* d_ws, size_t ws_size,
                              hipStream_t stream) {
  const float* z   = (const float*)d_in[0];   // [16,256,32,32] f32
  const float* emb = (const float*)d_in[1];   // [8192,256] f32
  float*       out = (float*)d_out;           // [16,256,32,32] f32
  float*       ehn = (float*)d_ws;            // 8192 f32 half-norms (32KB)

  ehn_kernel<<<K_CODES / 8, 256, 0, stream>>>(emb, ehn);
  vq_kernel<<<N_TOTAL / M_BLOCK, 256, 0, stream>>>(z, emb, ehn, out);
}
